// ConstraintLoss_17892833755337
// MI455X (gfx1250) — matrix operand order repaired and checked by reading, not compiled, not run on hardware
//
#include <hip/hip_runtime.h>
#include <math.h>

typedef __attribute__((ext_vector_type(2)))  float     v2f;
typedef __attribute__((ext_vector_type(8)))  float     v8f;
typedef __attribute__((ext_vector_type(8)))  _Float16  v8h;
typedef __attribute__((ext_vector_type(16))) _Float16  v16h;

#define NB 8
#define NT 1024
#define NC 256
#define NR 8
#define NTOK (NB*NT)       // 8192
#define INV_TAU (1.0f/0.07f)

// ---- workspace layout (bytes) ----
static constexpr size_t OFF_XHI  = 0;                                   // _Float16[NTOK*NC]  4 MB
static constexpr size_t OFF_XLO  = OFF_XHI  + (size_t)NTOK*NC*2;        // _Float16[NTOK*NC]  4 MB
static constexpr size_t OFF_CNT  = OFF_XLO  + (size_t)NTOK*NC*2;        // int[NB*NR]
static constexpr size_t OFF_RANK = OFF_CNT  + 64*4;                     // int[NTOK]
static constexpr size_t OFF_POOL = OFF_RANK + (size_t)NTOK*4;           // float[64*NC]
static constexpr size_t OFF_FS   = OFF_POOL + (size_t)64*NC*4;          // float[64*NC]
static constexpr size_t OFF_SS   = OFF_FS   + (size_t)64*NC*4;          // float[64*NC]
static constexpr size_t OFF_SCAL = OFF_SS   + (size_t)64*NC*4;          // float[4]: contra_sum, nvalid, anti, acyc

__device__ __forceinline__ float wred32(float x){
  #pragma unroll
  for(int m=16;m>=1;m>>=1) x += __shfl_xor(x, m, 32);
  return x;
}

__global__ void zero_k(float* scal){
  if (threadIdx.x < 4) scal[threadIdx.x] = 0.0f;
}

// one wave per token: L2-normalize each 256-dim embedding (eps 1e-12) and emit the
// split-precision pair: hi = f16(x), lo = f16(x - hi).  hi*hi'+hi*lo'+lo*hi' recovers
// the fp32 dot to ~1e-6, so the tau=0.07 exp amplification stays accurate.
__global__ void normalize_k(const float* __restrict__ emb,
                            _Float16* __restrict__ xhi, _Float16* __restrict__ xlo){
  int wave = threadIdx.x >> 5, lane = threadIdx.x & 31;
  int tok = blockIdx.x * 8 + wave;
  const float* src = emb + (size_t)tok * NC;
  float vals[8]; float s = 0.0f;
  #pragma unroll
  for (int k = 0; k < 8; ++k){ float v = src[lane + 32*k]; vals[k] = v; s += v*v; }
  s = wred32(s);
  float inv = 1.0f / fmaxf(sqrtf(s), 1e-12f);
  _Float16* dh = xhi + (size_t)tok * NC;
  _Float16* dl = xlo + (size_t)tok * NC;
  #pragma unroll
  for (int k = 0; k < 8; ++k){
    float v = vals[k] * inv;
    _Float16 h = (_Float16)v;
    dh[lane + 32*k] = h;
    dl[lane + 32*k] = (_Float16)(v - (float)h);
  }
}

// sequential per-batch rank within each (b, rel) chain + per-(b,r) counts
__global__ void rank_k(const int* __restrict__ rel, int* __restrict__ rank, int* __restrict__ cnt){
  int b = threadIdx.x;
  if (b >= NB) return;
  int ctr[NR];
  #pragma unroll
  for (int r = 0; r < NR; ++r) ctr[r] = 0;
  for (int t = 0; t < NT; ++t){
    int rr = rel[b*NT + t] & 7;
    rank[b*NT + t] = ctr[rr]++;
  }
  for (int r = 0; r < NR; ++r) cnt[b*NR + r] = ctr[r];
}

// per-(b,r): masked sums over T for pooled / first-half / second-half, per channel
__global__ void pool_k(const float* __restrict__ emb, const int* __restrict__ rel,
                       const int* __restrict__ rank, const int* __restrict__ cnt,
                       float* __restrict__ pooled, float* __restrict__ fsum, float* __restrict__ ssum){
  int br = blockIdx.x;            // b*8 + r
  int b = br >> 3, r = br & 7;
  int c = threadIdx.x;            // 0..255
  int half = cnt[br] >> 1;        // floor(n/2)
  float ps = 0.0f, fs = 0.0f, ss = 0.0f;
  for (int t = 0; t < NT; ++t){
    if (rel[b*NT + t] == r){
      float v = emb[((size_t)(b*NT + t))*NC + c];
      ps += v;
      if (rank[b*NT + t] < half) fs += v; else ss += v;
    }
  }
  pooled[(size_t)br*NC + c] = ps;
  fsum  [(size_t)br*NC + c] = fs;
  ssum  [(size_t)br*NC + c] = ss;
}

// one block: antisymmetry + acyclicity terms
__global__ void small_k(const int* __restrict__ cnt, const float* __restrict__ pooled,
                        const float* __restrict__ fsum, const float* __restrict__ ssum,
                        float* __restrict__ scal){
  __shared__ float sM[NR*NC];
  __shared__ float sNInv[NR];
  __shared__ int   sTot[NR];
  __shared__ float sPair[28];
  __shared__ int   sPairW[28];
  __shared__ float sAc;
  __shared__ int   sAcN;
  int tid = threadIdx.x, wave = tid >> 5, lane = tid & 31;
  if (tid == 0){ sAc = 0.0f; sAcN = 0; }
  if (tid < NR){
    int s = 0;
    for (int b = 0; b < NB; ++b) s += cnt[b*NR + tid];
    sTot[tid] = s;
  }
  // means[r][c] = mean_b( pooled_sum / max(cnt,1) )
  for (int idx = tid; idx < NR*NC; idx += 256){
    int r = idx >> 8, c = idx & 255;
    float m = 0.0f;
    for (int b = 0; b < NB; ++b)
      m += pooled[(size_t)(b*NR + r)*NC + c] / fmaxf((float)cnt[b*NR + r], 1.0f);
    sM[idx] = m * (1.0f/NB);
  }
  __syncthreads();
  { // row norms: wave r
    int r = wave;
    float s = 0.0f;
    for (int c = lane; c < NC; c += 32){ float v = sM[r*NC + c]; s += v*v; }
    s = wred32(s);
    if (lane == 0) sNInv[r] = 1.0f / fmaxf(sqrtf(s), 1e-12f);
  }
  __syncthreads();
  if (tid < 28){ // upper-triangular pair sims
    int i = 0, rem = tid;
    while (rem >= (NR - 1 - i)){ rem -= (NR - 1 - i); ++i; }
    int j = i + 1 + rem;
    float d = 0.0f;
    for (int c = 0; c < NC; ++c) d += sM[i*NC + c] * sM[j*NC + c];
    sPair[tid]  = d * sNInv[i] * sNInv[j];
    sPairW[tid] = ((sTot[i] > 0) && (sTot[j] > 0) && (i > 0)) ? 1 : 0;  // j>=1 always
  }
  // acyclicity: wave handles (b=wave, r=k)
  for (int k = 0; k < NR; ++k){
    int b = wave, r = k;
    int n = cnt[b*NR + r]; int half = n >> 1;
    float fd = fmaxf((float)half, 1.0f), sd = fmaxf((float)(n - half), 1.0f);
    float ff = 0.0f, s2 = 0.0f, fs2 = 0.0f;
    for (int c = lane; c < NC; c += 32){
      float f = fsum[(size_t)(b*NR + r)*NC + c] / fd;
      float s = ssum[(size_t)(b*NR + r)*NC + c] / sd;
      ff += f*f; s2 += s*s; fs2 += f*s;
    }
    ff = wred32(ff); s2 = wred32(s2); fs2 = wred32(fs2);
    if (lane == 0 && n >= 4 && r > 0){
      float sim = fs2 * (1.0f/fmaxf(sqrtf(ff), 1e-12f)) * (1.0f/fmaxf(sqrtf(s2), 1e-12f));
      atomicAdd(&sAc, sim);
      atomicAdd(&sAcN, 1);
    }
  }
  __syncthreads();
  if (tid == 0){
    float asum = 0.0f; int np = 0;
    for (int p = 0; p < 28; ++p){ if (sPairW[p]){ asum += sPair[p]; ++np; } }
    scal[2] = (np > 0)   ? asum / fmaxf((float)np, 1.0f)   * 0.2f : 0.0f;   // anti
    scal[3] = (sAcN > 0) ? sAc  / fmaxf((float)sAcN, 1.0f) * 0.2f : 0.0f;   // acyc
  }
}

// fused flash-style contrastive with split-precision f16 WMMA.
// 16-row tile per block (A hi/lo staged in LDS), 8 waves stream the 512 column tiles.
// Per K=32 chunk: D += Ahi*Bhi + Ahi*Blo + Alo*Bhi  (fp32 accumulate, error ~2^-22).
__global__ void __launch_bounds__(256) contra_k(const _Float16* __restrict__ xhi,
                                                const _Float16* __restrict__ xlo,
                                                const int* __restrict__ y,
                                                float* __restrict__ scal){
  __shared__ _Float16 Ahi[16*NC];    // 8 KB
  __shared__ _Float16 Alo[16*NC];    // 8 KB
  __shared__ int   yAs[16];
  __shared__ float denS[16], numS[16];
  int tid = threadIdx.x, wave = tid >> 5, lane = tid & 31;
  int i0 = blockIdx.x * 16;

  { // cooperative A-tile load: 512 float4 per array
    const float4* s1 = (const float4*)(xhi + (size_t)i0 * NC);
    const float4* s2 = (const float4*)(xlo + (size_t)i0 * NC);
    float4* d1 = (float4*)Ahi;
    float4* d2 = (float4*)Alo;
    for (int idx = tid; idx < 512; idx += 256){ d1[idx] = s1[idx]; d2[idx] = s2[idx]; }
  }
  if (tid < 16){ yAs[tid] = y[i0 + tid]; denS[tid] = 0.0f; numS[tid] = 0.0f; }
  __syncthreads();

  int h = lane >> 4, nl = lane & 15;
  int ym[8];
  #pragma unroll
  for (int v = 0; v < 8; ++v) ym[v] = yAs[v + 8*h];
  float denA[8], numA[8];
  #pragma unroll
  for (int v = 0; v < 8; ++v){ denA[v] = 0.0f; numA[v] = 0.0f; }

  // 16-bit A layout: lane half h, row m=nl; K range kb+16h..+15, consecutive
  const _Float16* aph = Ahi + nl*NC + 16*h;
  const _Float16* apl = Alo + nl*NC + 16*h;

  for (int jt = wave; jt < NTOK/16; jt += 8){
    int j0 = jt*16, ng = j0 + nl;
    int yn = y[ng];
    // 16-bit B layout: column n=nl, K range kb+16h..+15, consecutive in memory
    const _Float16* bph = xhi + (size_t)ng*NC + 16*h;
    const _Float16* bpl = xlo + (size_t)ng*NC + 16*h;
    v8f acc = {0.0f,0.0f,0.0f,0.0f,0.0f,0.0f,0.0f,0.0f};
    #pragma unroll
    for (int kc = 0; kc < 8; ++kc){            // K = 256 in chunks of 32
      int kb = 32*kc;
      v8h ah0 = *(const v8h*)(aph + kb);               // ds_load_b128
      v8h ah1 = *(const v8h*)(aph + kb + 8);
      v8h al0 = *(const v8h*)(apl + kb);
      v8h al1 = *(const v8h*)(apl + kb + 8);
      v16h a_hi = __builtin_shufflevector(ah0, ah1, 0,1,2,3,4,5,6,7,8,9,10,11,12,13,14,15);
      v16h a_lo = __builtin_shufflevector(al0, al1, 0,1,2,3,4,5,6,7,8,9,10,11,12,13,14,15);
      v16h b_hi = *(const v16h*)(bph + kb);            // 2x global_load_b128 (L2 resident)
      v16h b_lo = *(const v16h*)(bpl + kb);
      acc = __builtin_amdgcn_wmma_f32_16x16x32_f16(false, a_hi, false, b_hi, (short)0, acc, false, false);
      acc = __builtin_amdgcn_wmma_f32_16x16x32_f16(false, a_hi, false, b_lo, (short)0, acc, false, false);
      acc = __builtin_amdgcn_wmma_f32_16x16x32_f16(false, a_lo, false, b_hi, (short)0, acc, false, false);
    }
    bool cv = yn > 0;
    #pragma unroll
    for (int v = 0; v < 8; ++v){
      int mg = i0 + v + 8*h;                  // D layout: row = vgpr + 8*half, col = lane&15
      float e = expf(acc[v] * INV_TAU);       // logits <= 14.3, safe in fp32, no max-trick needed
      bool ok = cv && (ng != mg);
      denA[v] += ok ? e : 0.0f;
      numA[v] += (ok && (yn == ym[v])) ? e : 0.0f;
    }
  }
  // reduce over the 16 column-lanes (stays within each lane-half)
  #pragma unroll
  for (int v = 0; v < 8; ++v){
    #pragma unroll
    for (int m = 8; m >= 1; m >>= 1){
      denA[v] += __shfl_xor(denA[v], m, 32);
      numA[v] += __shfl_xor(numA[v], m, 32);
    }
  }
  if (nl == 0){
    #pragma unroll
    for (int v = 0; v < 8; ++v){
      atomicAdd(&denS[v + 8*h], denA[v]);
      atomicAdd(&numS[v + 8*h], numA[v]);
    }
  }
  __syncthreads();
  if (tid < 16 && yAs[tid] > 0){
    float den = fmaxf(denS[tid], 1e-6f);
    float num = fmaxf(numS[tid], 1e-6f);
    atomicAdd(&scal[0], logf(den) - logf(num));   // -log(num/den)
    atomicAdd(&scal[1], 1.0f);                    // nvalid
  }
}

__global__ void final_k(const float* __restrict__ scal, float* __restrict__ out){
  if (threadIdx.x == 0){
    float contra = scal[0] / fmaxf(scal[1], 1.0f);
    out[0] = scal[2];                       // anti
    out[1] = scal[3];                       // acyc
    out[2] = contra;                        // contra
    out[3] = scal[2] + scal[3] + contra;    // total
  }
}

extern "C" void kernel_launch(void* const* d_in, const int* in_sizes, int n_in,
                              void* d_out, int out_size, void* d_ws, size_t ws_size,
                              hipStream_t stream){
  (void)in_sizes; (void)n_in; (void)out_size; (void)ws_size;
  const float* emb = (const float*)d_in[0];
  const int*   rel = (const int*)d_in[1];
  char* ws = (char*)d_ws;
  _Float16* xhi  = (_Float16*)(ws + OFF_XHI);
  _Float16* xlo  = (_Float16*)(ws + OFF_XLO);
  int*   cnt    = (int*)  (ws + OFF_CNT);
  int*   rank   = (int*)  (ws + OFF_RANK);
  float* pooled = (float*)(ws + OFF_POOL);
  float* fsum   = (float*)(ws + OFF_FS);
  float* ssum   = (float*)(ws + OFF_SS);
  float* scal   = (float*)(ws + OFF_SCAL);
  float* out    = (float*)d_out;

  hipLaunchKernelGGL(zero_k,      dim3(1),       dim3(64),  0, stream, scal);
  hipLaunchKernelGGL(normalize_k, dim3(NTOK/8),  dim3(256), 0, stream, emb, xhi, xlo);
  hipLaunchKernelGGL(rank_k,      dim3(1),       dim3(8),   0, stream, rel, rank, cnt);
  hipLaunchKernelGGL(pool_k,      dim3(NB*NR),   dim3(NC),  0, stream, emb, rel, rank, cnt, pooled, fsum, ssum);
  hipLaunchKernelGGL(small_k,     dim3(1),       dim3(256), 0, stream, cnt, pooled, fsum, ssum, scal);
  hipLaunchKernelGGL(contra_k,    dim3(NTOK/16), dim3(256), 0, stream, xhi, xlo, rel, scal);
  hipLaunchKernelGGL(final_k,     dim3(1),       dim3(1),   0, stream, scal, out);
}
